// HybridAttention_55834574848492
// MI455X (gfx1250) — matrix operand order repaired and checked by reading, not compiled, not run on hardware
//
#include <hip/hip_runtime.h>

#define EMBED   1024
#define HEADS   16
#define HDIM    64
#define SEQ     2048
#define BATCH   4
#define WINDOW  256
#define NGLOBAL 4

typedef __attribute__((ext_vector_type(16))) __bf16 v16bf;
typedef __attribute__((ext_vector_type(8)))  float  v8f;
typedef __attribute__((ext_vector_type(4)))  int    v4i;

union Frag {
  v16bf v;
  uint4 q[2];
};

__device__ __forceinline__ unsigned short f2bf(float f) {
  union { float f; unsigned int u; } cv; cv.f = f;
  unsigned int u = cv.u;
  u += 0x7FFFu + ((u >> 16) & 1u);   // round-to-nearest-even
  return (unsigned short)(u >> 16);
}

__device__ __forceinline__ unsigned int pk2(float a, float b) {
  return (unsigned int)f2bf(a) | ((unsigned int)f2bf(b) << 16);
}

__device__ __forceinline__ v8f wmma_bf16(v16bf a, v16bf b, v8f c) {
#if defined(__gfx1250__)
  // D = A(16x32 bf16) * B(32x16 bf16) + C(16x16 f32)
  return __builtin_amdgcn_wmma_f32_16x16x32_bf16(false, a, false, b, (short)0, c,
                                                 false, false);
#else
  return c;
#endif
}

__device__ __forceinline__ void sched_fence() {
#if defined(__HIP_DEVICE_COMPILE__)
  __builtin_amdgcn_wave_barrier();
#endif
}

// ---- CDNA5 async global->LDS (ASYNCcnt-tracked DMA into LDS) --------------
#if defined(__gfx1250__) && defined(__HIP_DEVICE_COMPILE__)
#define HAS_ASYNC_LDS __has_builtin(__builtin_amdgcn_global_load_async_to_lds_b128)
#else
#define HAS_ASYNC_LDS 0
#endif

__device__ __forceinline__ void async_b128(const unsigned short* g,
                                           unsigned short* l) {
#if HAS_ASYNC_LDS
  __builtin_amdgcn_global_load_async_to_lds_b128(
      (__attribute__((address_space(1))) v4i*)(v4i*)(void*)g,
      (__attribute__((address_space(3))) v4i*)(v4i*)(void*)l, 0, 0);
#else
  *(uint4*)l = *(const uint4*)g;   // fallback: VGPR-staged copy
#endif
}

#if HAS_ASYNC_LDS
#if __has_builtin(__builtin_amdgcn_s_wait_asynccnt)
#define WAIT_ASYNC(n) __builtin_amdgcn_s_wait_asynccnt(n)
#else
#define WAIT_ASYNC(n) asm volatile("s_wait_asynccnt %0" ::"i"(n) : "memory")
#endif
#else
#define WAIT_ASYNC(n) do {} while (0)
#endif

// Stage one 128x32 A-tile + 128x32 B-tile (bf16) into LDS, 4 b128/thread.
__device__ __forceinline__ void stage_tile(const unsigned short* __restrict__ A,
                                           const unsigned short* __restrict__ B,
                                           unsigned short* As, unsigned short* Bs,
                                           int tid, int m0, int n0, int k0) {
#pragma unroll
  for (int i = 0; i < 2; ++i) {
    int c = tid * 2 + i;               // 0..511 chunk id (8 halves each)
    int row = c >> 2;
    int off = (c & 3) * 8;
    async_b128(A + (size_t)(m0 + row) * EMBED + k0 + off, As + row * 32 + off);
    async_b128(B + (size_t)(n0 + row) * EMBED + k0 + off, Bs + row * 32 + off);
  }
}

// ---------------------------------------------------------------------------
// Elementwise f32 -> bf16 (x), vectorized 8 elements / thread.
// ---------------------------------------------------------------------------
__global__ void convert_f32_bf16(const float* __restrict__ src,
                                 unsigned short* __restrict__ dst, int n8) {
  int idx = blockIdx.x * blockDim.x + threadIdx.x;
  if (idx >= n8) return;
  const float4* s = (const float4*)src + (size_t)idx * 2;
  float4 f0 = s[0], f1 = s[1];
  uint4 p;
  p.x = pk2(f0.x, f0.y); p.y = pk2(f0.z, f0.w);
  p.z = pk2(f1.x, f1.y); p.w = pk2(f1.z, f1.w);
  ((uint4*)dst)[idx] = p;
}

// dst[n*K + k] = bf16(src[k*N + n])   (one-shot weight transpose+convert)
__global__ void transpose_to_bf16(const float* __restrict__ src,
                                  unsigned short* __restrict__ dst,
                                  int K, int N) {
  int idx = blockIdx.x * blockDim.x + threadIdx.x;
  if (idx >= K * N) return;
  int n = idx / K;
  int k = idx - n * K;
  dst[idx] = f2bf(src[(size_t)k * N + n]);
}

// ---------------------------------------------------------------------------
// QKV GEMM: C[8192,3072] = xbf[8192,1024] * wT[3072,1024]^T + b
// Async double-buffered LDS pipeline; epilogue scatters bf16 into
// Q[B,H,S,D], K[B,H,S,D], Vt[B,H,D,S]. `which` is uniform per block.
// ---------------------------------------------------------------------------
__global__ __launch_bounds__(256) void gemm_qkv_kernel(
    const unsigned short* __restrict__ xbf, const unsigned short* __restrict__ wT,
    const float* __restrict__ bias,
    unsigned short* __restrict__ Qb, unsigned short* __restrict__ Kb,
    unsigned short* __restrict__ Vt) {
  __shared__ unsigned short As[2][128 * 32];
  __shared__ unsigned short Bs[2][128 * 32];
  const int tid  = threadIdx.x;
  const int lane = tid & 31, wave = tid >> 5;
  const int hi = lane >> 4, ln = lane & 15;
  const int wm = wave >> 1, wn = wave & 1;
  const int m0 = blockIdx.y * 128;
  const int n0 = blockIdx.x * 128;

  v8f acc[2][4];
#pragma unroll
  for (int i = 0; i < 2; ++i)
#pragma unroll
    for (int j = 0; j < 4; ++j)
#pragma unroll
      for (int r = 0; r < 8; ++r) acc[i][j][r] = 0.f;

  const int NT = EMBED / 32;                 // 32 K-steps
  stage_tile(xbf, wT, As[0], Bs[0], tid, m0, n0, 0);
  for (int t = 0; t < NT; ++t) {
    const int cur = t & 1;
    if (t + 1 < NT) {
      stage_tile(xbf, wT, As[1 - cur], Bs[1 - cur], tid, m0, n0, (t + 1) * 32);
      WAIT_ASYNC(4);                         // tile t complete; t+1 in flight
    } else {
      WAIT_ASYNC(0);
    }
    __syncthreads();
    const unsigned short* Ab = As[cur];
    const unsigned short* Bb = Bs[cur];
    Frag a[2], b[4];
#pragma unroll
    for (int i = 0; i < 2; ++i) {
      int row = wm * 32 + i * 16 + ln;
      a[i].q[0] = *(const uint4*)&Ab[row * 32 + hi * 16];
      a[i].q[1] = *(const uint4*)&Ab[row * 32 + hi * 16 + 8];
    }
#pragma unroll
    for (int j = 0; j < 4; ++j) {
      int row = wn * 64 + j * 16 + ln;
      b[j].q[0] = *(const uint4*)&Bb[row * 32 + hi * 16];
      b[j].q[1] = *(const uint4*)&Bb[row * 32 + hi * 16 + 8];
    }
#pragma unroll
    for (int i = 0; i < 2; ++i)
#pragma unroll
      for (int j = 0; j < 4; ++j)
        acc[i][j] = wmma_bf16(a[i].v, b[j].v, acc[i][j]);
    __syncthreads();
  }

  // Epilogue: q/k/v selector is uniform for the whole 128-wide N block.
  const int which = n0 >> 10;
  const int e0 = n0 & 1023;
#pragma unroll
  for (int i = 0; i < 2; ++i) {
    int mbase = m0 + wm * 32 + i * 16;
    int bb = mbase >> 11;                    // batch (SEQ = 2048)
    int srow = mbase & (SEQ - 1);
#pragma unroll
    for (int j = 0; j < 4; ++j) {
      int e = e0 + wn * 64 + j * 16 + ln;    // 0..1023 within q/k/v
      int h = e >> 6;
      int d = e & 63;
      float bv = bias[which * EMBED + e];
      size_t hb = (size_t)(bb * HEADS + h);
      if (which == 0) {
#pragma unroll
        for (int r = 0; r < 8; ++r) {
          int s = srow + r + 8 * hi;
          Qb[(hb * SEQ + s) * HDIM + d] = f2bf(acc[i][j][r] + bv);
        }
      } else if (which == 1) {
#pragma unroll
        for (int r = 0; r < 8; ++r) {
          int s = srow + r + 8 * hi;
          Kb[(hb * SEQ + s) * HDIM + d] = f2bf(acc[i][j][r] + bv);
        }
      } else {
#pragma unroll
        for (int r = 0; r < 8; ++r) {
          int s = srow + r + 8 * hi;
          Vt[(hb * HDIM + d) * SEQ + s] = f2bf(acc[i][j][r] + bv);
        }
      }
    }
  }
}

// ---------------------------------------------------------------------------
// Flash attention: 1 wave = one 16-query tile of one (b,h); 32-key steps.
// Heads 0..3: full causal. Heads 4..15: causal window of WINDOW.
// ---------------------------------------------------------------------------
__global__ __launch_bounds__(256) void attn_kernel(
    const unsigned short* __restrict__ Qb, const unsigned short* __restrict__ Kb,
    const unsigned short* __restrict__ Vt, unsigned short* __restrict__ ctx) {
  __shared__ unsigned short Pl[8][16 * 32];   // per-wave P transpose staging
  const int tid  = threadIdx.x;
  const int lane = tid & 31, wave = tid >> 5;
  const int hi = lane >> 4, ln = lane & 15;
  int blk = blockIdx.x;                 // B*H*(SEQ/128) blocks
  int qb = blk & 15;                    // SEQ/128 = 16
  int bh = blk >> 4;
  int h = bh & (HEADS - 1);
  int b = bh >> 4;
  int q0 = qb * 128 + wave * 16;

  const size_t head_off = (size_t)(b * HEADS + h) * SEQ * HDIM;
  const unsigned short* Qh = Qb + head_off;   // [S,D]
  const unsigned short* Kh = Kb + head_off;   // [S,D]
  const unsigned short* Vh = Vt + head_off;   // [D,S]

  // Q fragments (16 queries x 64 dims -> two 16x32 A-frags), loaded once.
  Frag qa[2];
  {
    const unsigned short* qrow = Qh + (size_t)(q0 + ln) * HDIM;
#pragma unroll
    for (int dh = 0; dh < 2; ++dh) {
      qa[dh].q[0] = *(const uint4*)(qrow + dh * 32 + hi * 16);
      qa[dh].q[1] = *(const uint4*)(qrow + dh * 32 + hi * 16 + 8);
    }
  }

  v8f o[4];
  float Mrow[8], Lrow[8];
#pragma unroll
  for (int j = 0; j < 4; ++j)
#pragma unroll
    for (int r = 0; r < 8; ++r) o[j][r] = 0.f;
#pragma unroll
  for (int r = 0; r < 8; ++r) { Mrow[r] = -1e30f; Lrow[r] = 0.f; }

  const bool isGlobal = (h < NGLOBAL);
  const int kend = q0 + 16;             // exclusive causal bound
  int kstart = 0;
  if (!isGlobal) {
    kstart = q0 - WINDOW;
    if (kstart < 0) kstart = 0;
    kstart &= ~31;
  }
  unsigned short* P = Pl[wave];

  for (int k0 = kstart; k0 < kend; k0 += 32) {
    // ---- scores: S(16x32) = Q(16x64) * K^T  (4 WMMAs) ----
    v8f sc[2];
#pragma unroll
    for (int kt = 0; kt < 2; ++kt)
#pragma unroll
      for (int r = 0; r < 8; ++r) sc[kt][r] = 0.f;
#pragma unroll
    for (int kt = 0; kt < 2; ++kt) {
      const unsigned short* krow = Kh + (size_t)(k0 + kt * 16 + ln) * HDIM;
#pragma unroll
      for (int dh = 0; dh < 2; ++dh) {
        Frag kb;
        kb.q[0] = *(const uint4*)(krow + dh * 32 + hi * 16);
        kb.q[1] = *(const uint4*)(krow + dh * 32 + hi * 16 + 8);
        sc[kt] = wmma_bf16(qa[dh].v, kb.v, sc[kt]);
      }
    }
    // ---- masked online softmax (row r+8*hi lives in VGPR r, col in lane%16)
    float alpha[8];
#pragma unroll
    for (int r = 0; r < 8; ++r) {
      int q = q0 + r + 8 * hi;
      float mx = -1e30f;
#pragma unroll
      for (int kt = 0; kt < 2; ++kt) {
        int k = k0 + kt * 16 + ln;
        int diff = q - k;
        float v = sc[kt][r] * 0.125f;   // 1/sqrt(64)
        bool masked = (diff < 0) || (!isGlobal && diff > WINDOW);
        v = masked ? -1e30f : v;
        sc[kt][r] = v;
        mx = fmaxf(mx, v);
      }
#pragma unroll
      for (int m = 1; m < 16; m <<= 1) mx = fmaxf(mx, __shfl_xor(mx, m, 32));
      float Mnew = fmaxf(Mrow[r], mx);
      alpha[r] = __expf(Mrow[r] - Mnew);
      Mrow[r] = Mnew;
      float sum = 0.f;
#pragma unroll
      for (int kt = 0; kt < 2; ++kt) {
        float p = __expf(sc[kt][r] - Mnew);
        sc[kt][r] = p;
        sum += p;
      }
#pragma unroll
      for (int m = 1; m < 16; m <<= 1) sum += __shfl_xor(sum, m, 32);
      Lrow[r] = Lrow[r] * alpha[r] + sum;
    }
#pragma unroll
    for (int j = 0; j < 4; ++j)
#pragma unroll
      for (int r = 0; r < 8; ++r) o[j][r] *= alpha[r];

    // ---- transpose P (C-layout -> A-layout) through LDS ----
#pragma unroll
    for (int kt = 0; kt < 2; ++kt)
#pragma unroll
      for (int r = 0; r < 8; ++r)
        P[(r + 8 * hi) * 32 + kt * 16 + ln] = f2bf(sc[kt][r]);
    sched_fence();   // same-wave LDS ops are in-order; keep compiler honest
    Frag pa;
    pa.q[0] = *(const uint4*)&P[ln * 32 + hi * 16];
    pa.q[1] = *(const uint4*)&P[ln * 32 + hi * 16 + 8];

    // ---- O(16x64) += P(16x32) * V(32x64)  (4 WMMAs, V pre-transposed) ----
#pragma unroll
    for (int j = 0; j < 4; ++j) {
      const unsigned short* vrow = Vh + (size_t)(j * 16 + ln) * SEQ + k0 + hi * 16;
      Frag vb;
      vb.q[0] = *(const uint4*)(vrow);
      vb.q[1] = *(const uint4*)(vrow + 8);
      o[j] = wmma_bf16(pa.v, vb.v, o[j]);
    }
  }

  // ---- normalize and emit ctx bf16 [B,S,E] (heads concatenated) ----
#pragma unroll
  for (int j = 0; j < 4; ++j) {
    int d = j * 16 + ln;
#pragma unroll
    for (int r = 0; r < 8; ++r) {
      int s = q0 + r + 8 * hi;
      float val = o[j][r] / Lrow[r];
      ctx[((size_t)b * SEQ + s) * EMBED + h * HDIM + d] = f2bf(val);
    }
  }
}

// ---------------------------------------------------------------------------
// Output GEMM: out[8192,1024] = ctx_bf16[8192,1024] * w_outT[1024,1024]^T + b
// Same async double-buffered pipeline; f32 row-major epilogue.
// ---------------------------------------------------------------------------
__global__ __launch_bounds__(256) void gemm_out_kernel(
    const unsigned short* __restrict__ ctx, const unsigned short* __restrict__ wT,
    const float* __restrict__ bias, float* __restrict__ out) {
  __shared__ unsigned short As[2][128 * 32];
  __shared__ unsigned short Bs[2][128 * 32];
  const int tid  = threadIdx.x;
  const int lane = tid & 31, wave = tid >> 5;
  const int hi = lane >> 4, ln = lane & 15;
  const int wm = wave >> 1, wn = wave & 1;
  const int m0 = blockIdx.y * 128;
  const int n0 = blockIdx.x * 128;

  v8f acc[2][4];
#pragma unroll
  for (int i = 0; i < 2; ++i)
#pragma unroll
    for (int j = 0; j < 4; ++j)
#pragma unroll
      for (int r = 0; r < 8; ++r) acc[i][j][r] = 0.f;

  const int NT = EMBED / 32;
  stage_tile(ctx, wT, As[0], Bs[0], tid, m0, n0, 0);
  for (int t = 0; t < NT; ++t) {
    const int cur = t & 1;
    if (t + 1 < NT) {
      stage_tile(ctx, wT, As[1 - cur], Bs[1 - cur], tid, m0, n0, (t + 1) * 32);
      WAIT_ASYNC(4);
    } else {
      WAIT_ASYNC(0);
    }
    __syncthreads();
    const unsigned short* Ab = As[cur];
    const unsigned short* Bb = Bs[cur];
    Frag a[2], b[4];
#pragma unroll
    for (int i = 0; i < 2; ++i) {
      int row = wm * 32 + i * 16 + ln;
      a[i].q[0] = *(const uint4*)&Ab[row * 32 + hi * 16];
      a[i].q[1] = *(const uint4*)&Ab[row * 32 + hi * 16 + 8];
    }
#pragma unroll
    for (int j = 0; j < 4; ++j) {
      int row = wn * 64 + j * 16 + ln;
      b[j].q[0] = *(const uint4*)&Bb[row * 32 + hi * 16];
      b[j].q[1] = *(const uint4*)&Bb[row * 32 + hi * 16 + 8];
    }
#pragma unroll
    for (int i = 0; i < 2; ++i)
#pragma unroll
      for (int j = 0; j < 4; ++j)
        acc[i][j] = wmma_bf16(a[i].v, b[j].v, acc[i][j]);
    __syncthreads();
  }

#pragma unroll
  for (int i = 0; i < 2; ++i) {
    int mbase = m0 + wm * 32 + i * 16;
#pragma unroll
    for (int j = 0; j < 4; ++j) {
      int n = n0 + wn * 64 + j * 16 + ln;
      float bv = bias[n];
#pragma unroll
      for (int r = 0; r < 8; ++r)
        out[(size_t)(mbase + r + 8 * hi) * EMBED + n] = acc[i][j][r] + bv;
    }
  }
}

// ---------------------------------------------------------------------------
extern "C" void kernel_launch(void* const* d_in, const int* in_sizes, int n_in,
                              void* d_out, int out_size, void* d_ws, size_t ws_size,
                              hipStream_t stream) {
  const float* x     = (const float*)d_in[0];  // [4,2048,1024]
  const float* w_qkv = (const float*)d_in[1];  // [1024,3072]
  const float* b_qkv = (const float*)d_in[2];  // [3072]
  const float* w_out = (const float*)d_in[3];  // [1024,1024]
  const float* b_out = (const float*)d_in[4];  // [1024]
  float* out = (float*)d_out;

  // Workspace layout (16B-aligned; total = 88 MiB)
  char* ws = (char*)d_ws;
  unsigned short* xbf   = (unsigned short*)ws; ws += (size_t)BATCH * SEQ * EMBED * 2;
  unsigned short* wqkvT = (unsigned short*)ws; ws += (size_t)3 * EMBED * EMBED * 2;
  unsigned short* woutT = (unsigned short*)ws; ws += (size_t)EMBED * EMBED * 2;
  unsigned short* Qb    = (unsigned short*)ws; ws += (size_t)BATCH * SEQ * EMBED * 2;
  unsigned short* Kb    = (unsigned short*)ws; ws += (size_t)BATCH * SEQ * EMBED * 2;
  unsigned short* Vt    = (unsigned short*)ws; ws += (size_t)BATCH * SEQ * EMBED * 2;
  unsigned short* ctx   = (unsigned short*)ws; ws += (size_t)BATCH * SEQ * EMBED * 2;

  const int n8 = BATCH * SEQ * EMBED / 8;
  convert_f32_bf16<<<(n8 + 255) / 256, 256, 0, stream>>>(x, xbf, n8);
  transpose_to_bf16<<<(3 * EMBED * EMBED + 255) / 256, 256, 0, stream>>>(
      w_qkv, wqkvT, EMBED, 3 * EMBED);
  transpose_to_bf16<<<(EMBED * EMBED + 255) / 256, 256, 0, stream>>>(
      w_out, woutT, EMBED, EMBED);

  dim3 g1(3 * EMBED / 128, BATCH * SEQ / 128);
  gemm_qkv_kernel<<<g1, 256, 0, stream>>>(xbf, wqkvT, b_qkv, Qb, Kb, Vt);

  attn_kernel<<<BATCH * HEADS * (SEQ / 128), 256, 0, stream>>>(Qb, Kb, Vt, ctx);

  dim3 g2(EMBED / 128, BATCH * SEQ / 128);
  gemm_out_kernel<<<g2, 256, 0, stream>>>(ctx, woutT, b_out, out);
}
